// Model_11244224381161
// MI455X (gfx1250) — matrix operand order repaired and checked by reading, not compiled, not run on hardware
//
#include <hip/hip_runtime.h>
#include <math.h>

// ---------------------------------------------------------------------------
// Types for CDNA5 WMMA (wave32): v_wmma_f32_16x16x32_bf16
// ---------------------------------------------------------------------------
typedef __bf16 bf16;
typedef __attribute__((ext_vector_type(16))) __bf16 v16bf;
typedef __attribute__((ext_vector_type(8)))  float  v8f;

#define DEVINL __device__ __forceinline__

DEVINL bf16 f2bf(float f) {
  // round-to-nearest-even f32 -> bf16
  unsigned int x = __builtin_bit_cast(unsigned int, f);
  unsigned int r = (x + 0x7fffu + ((x >> 16) & 1u)) >> 16;
  unsigned short s = (unsigned short)r;
  return __builtin_bit_cast(bf16, s);
}

DEVINL v8f wmma_bf16(v16bf a, v16bf b, v8f c) {
  // D = A(16x32 bf16) x B(32x16 bf16) + C(16x16 f32)
  return __builtin_amdgcn_wmma_f32_16x16x32_bf16(
      /*neg_a=*/false, a, /*neg_b=*/false, b,
      /*c_mod=*/(short)0, c, /*reuse_a=*/false, /*reuse_b=*/false);
}

// A fragment (16x32, row-major source, bf16) from LDS.
// Lane l (l<16): M=l, K in {0..7,16..23}; lane l>=16: M=l-16, K in {8..15,24..31}
DEVINL v16bf load_a_bf(const bf16* __restrict__ A, int lda, int kb, int lane) {
  int m = lane & 15;
  int khalf = (lane < 16) ? 0 : 8;
  const bf16* row = A + m * lda + kb * 32;
  v16bf a;
#pragma unroll
  for (int v = 0; v < 8; ++v) {
    int k0 = ((v < 4) ? 0 : 16) + khalf + 2 * (v & 3);
    a[2 * v]     = row[k0];
    a[2 * v + 1] = row[k0 + 1];
  }
  return a;
}

// Same A fragment but sourced from f32 global memory (dense layers).
DEVINL v16bf load_a_f32(const float* __restrict__ A, int lda, int kb, int lane) {
  int m = lane & 15;
  int khalf = (lane < 16) ? 0 : 8;
  const float* row = A + (size_t)m * lda + kb * 32;
  v16bf a;
#pragma unroll
  for (int v = 0; v < 8; ++v) {
    int k0 = ((v < 4) ? 0 : 16) + khalf + 2 * (v & 3);
    a[2 * v]     = f2bf(row[k0]);
    a[2 * v + 1] = f2bf(row[k0 + 1]);
  }
  return a;
}

// B fragments are pre-packed so each lane's 16 bf16 values are contiguous:
// one 32-byte region -> two global_load_b128 per fragment (L2-resident weights).
DEVINL v16bf load_b_frag(const bf16* B, int numKb, int kb, int nt, int lane) {
  return *(const v16bf*)(B + ((size_t)(nt * numKb + kb) * 512) + lane * 16);
}

DEVINL float sigf(float x) { return 1.f / (1.f + __expf(-x)); }

// ---------------------------------------------------------------------------
// Prep: pack f32 weights -> padded, gate-split, WMMA-B-fragment bf16 layout
// dst logical [Kp][gN*gDstW]; fragment order: (nt*numKb + kb)*512 + lane*16 + elem
// ---------------------------------------------------------------------------
__global__ void k_pack(const float* __restrict__ src, bf16* __restrict__ dst,
                       int Ksrc, int Kp, int gN, int gSrcW, int gDstW) {
  int numKb = Kp >> 5;
  int Np = gN * gDstW;
  int total = Kp * Np;
  int srcW = gN * gSrcW;
  for (int e = blockIdx.x * blockDim.x + threadIdx.x; e < total;
       e += gridDim.x * blockDim.x) {
    int frag = e >> 9;
    int within = e & 511;
    int lane = within >> 4;
    int elem = within & 15;
    int v = elem >> 1, lo = elem & 1;
    int nt = frag / numKb, kb = frag - nt * numKb;
    int n_p = nt * 16 + (lane & 15);
    int k_p = kb * 32 + ((v < 4) ? 0 : 16) + ((lane < 16) ? 0 : 8) + 2 * (v & 3) + lo;
    int g = n_p / gDstW, cc = n_p - g * gDstW;
    float val = 0.f;
    if (k_p < Ksrc && cc < gSrcW && g < gN)
      val = src[(size_t)k_p * srcW + g * gSrcW + cc];
    dst[e] = f2bf(val);
  }
}

__global__ void k_packbias(const float* __restrict__ s, float* __restrict__ d,
                           int n, int np) {
  int i = blockIdx.x * blockDim.x + threadIdx.x;
  if (i < np) d[i] = (i < n) ? s[i] : 0.f;
}

__global__ void k_zero(float* __restrict__ p, int n) {
  for (int i = blockIdx.x * blockDim.x + threadIdx.x; i < n;
       i += gridDim.x * blockDim.x)
    p[i] = 0.f;
}

// ---------------------------------------------------------------------------
// Feature GRU: persistent per-batch-tile kernel. 13 blocks x 512 thr (16 waves).
// Wave w owns output columns [16w,16w+16) of each gate: accumulators cz/cr/cxh/crh
// stay in VGPRs; h fragment stays in VGPRs; bf16 copy of h in LDS feeds the
// recurrent WMMA of the next step. An opaque zero offset (empty asm) makes the
// weight addresses loop-variant so LICM cannot hoist 36 fragments into VGPRs
// (round-1 spill disaster) while keeping plain cacheable global_load_b128s.
// ---------------------------------------------------------------------------
__global__ void __launch_bounds__(512)
k_gru_feat(const float* __restrict__ x, const bf16* __restrict__ Wp,
           const bf16* __restrict__ Up, const float* __restrict__ b,
           float* __restrict__ hf) {
  __shared__ bf16 sh_x[16 * 128];
  __shared__ bf16 sh_h[16 * 256];
  const int bt = blockIdx.x;
  const int tid = threadIdx.x;
  const int w = tid >> 5, lane = tid & 31, c16 = lane & 15;

  for (int e = tid; e < 16 * 256; e += 512) sh_h[e] = f2bf(0.f);

  const int nz = 16 * w + c16;
  const float bz  = b[nz]        + b[768 + nz];
  const float br  = b[256 + nz]  + b[768 + 256 + nz];
  const float bxh = b[512 + nz];
  const float brh = b[768 + 512 + nz];

  v8f hfrag;
#pragma unroll
  for (int i = 0; i < 8; ++i) hfrag[i] = 0.f;

  for (int t = 0; t < 2048; ++t) {
    // opaque zero: loop-variant to the optimizer, no instructions emitted
    int opq = 0;
    asm volatile("" : "+s"(opq));
    const bf16* Wt = Wp + opq;
    const bf16* Ut = Up + opq;

    // stage x_t (f32 -> bf16) ; prefetch x_{t+1}
    for (int e = tid; e < 16 * 128; e += 512) {
      int r = e >> 7, k = e & 127;
      int gb = bt * 16 + r;
      float v = 0.f;
      if (gb < 200) {
        const float* p = x + ((size_t)gb * 2048 + t) * 128 + k;
        v = *p;
        if (t + 1 < 2048) __builtin_prefetch(p + 128, 0, 0);
      }
      sh_x[e] = f2bf(v);
    }
    __syncthreads();

    v8f cz, cr, cxh, crh;
#pragma unroll
    for (int i = 0; i < 8; ++i) { cz[i] = bz; cr[i] = br; cxh[i] = bxh; crh[i] = brh; }

#pragma unroll
    for (int kb = 0; kb < 4; ++kb) {  // x_t @ W  (K=128)
      v16bf a = load_a_bf(sh_x, 128, kb, lane);
      cz  = wmma_bf16(a, load_b_frag(Wt, 4, kb, w,      lane), cz);
      cr  = wmma_bf16(a, load_b_frag(Wt, 4, kb, 16 + w, lane), cr);
      cxh = wmma_bf16(a, load_b_frag(Wt, 4, kb, 32 + w, lane), cxh);
    }
#pragma unroll
    for (int kb = 0; kb < 8; ++kb) {  // h @ U  (K=256)
      v16bf a = load_a_bf(sh_h, 256, kb, lane);
      cz  = wmma_bf16(a, load_b_frag(Ut, 8, kb, w,      lane), cz);
      cr  = wmma_bf16(a, load_b_frag(Ut, 8, kb, 16 + w, lane), cr);
      crh = wmma_bf16(a, load_b_frag(Ut, 8, kb, 32 + w, lane), crh);
    }
    __syncthreads();

#pragma unroll
    for (int i = 0; i < 8; ++i) {  // GRU gates, all in registers
      float z  = sigf(cz[i]);
      float r  = sigf(cr[i]);
      float hh = tanhf(cxh[i] + r * crh[i]);
      float h  = z * hfrag[i] + (1.f - z) * hh;
      hfrag[i] = h;
      int row = (lane < 16) ? i : i + 8;
      sh_h[row * 256 + 16 * w + c16] = f2bf(h);
    }
  }
#pragma unroll
  for (int i = 0; i < 8; ++i) {
    int row = (lane < 16) ? i : i + 8;
    hf[(size_t)(bt * 16 + row) * 256 + 16 * w + c16] = hfrag[i];
  }
}

// ---------------------------------------------------------------------------
// Char GRU: H=200 padded to 208 (padded cols self-zero: zero weights+bias
// => z=0.5, hh=0 => h stays 0). 13 blocks x 416 thr (13 waves).
// ---------------------------------------------------------------------------
__global__ void __launch_bounds__(416)
k_gru_char(const int* __restrict__ ci, const float* __restrict__ emb,
           const bf16* __restrict__ Wp, const bf16* __restrict__ Up,
           const float* __restrict__ b, float* __restrict__ hc) {
  __shared__ bf16 sh_x[16 * 160];   // EMB=150 padded to 160
  __shared__ bf16 sh_h[16 * 224];   // H=200 padded to 224 (7 k-blocks)
  const int bt = blockIdx.x;
  const int tid = threadIdx.x;
  const int w = tid >> 5, lane = tid & 31, c16 = lane & 15;

  for (int e = tid; e < 16 * 224; e += 416) sh_h[e] = f2bf(0.f);

  const int col = 16 * w + c16;           // 0..207
  const bool cv = (col < 200);
  const float bz  = cv ? (b[col]       + b[600 + col])       : 0.f;
  const float br  = cv ? (b[200 + col] + b[600 + 200 + col]) : 0.f;
  const float bxh = cv ? b[400 + col]        : 0.f;
  const float brh = cv ? b[600 + 400 + col]  : 0.f;

  v8f hfrag;
#pragma unroll
  for (int i = 0; i < 8; ++i) hfrag[i] = 0.f;

  for (int t = 0; t < 128; ++t) {
    int opq = 0;
    asm volatile("" : "+s"(opq));
    const bf16* Wt = Wp + opq;
    const bf16* Ut = Up + opq;

    // embedding gather + stage
    for (int e = tid; e < 16 * 160; e += 416) {
      int r = e / 160, k = e - r * 160;
      int gb = bt * 16 + r;
      float v = 0.f;
      if (gb < 200 && k < 150) {
        int idx = ci[gb * 128 + t];
        v = emb[(size_t)idx * 150 + k];
      }
      sh_x[e] = f2bf(v);
    }
    __syncthreads();

    v8f cz, cr, cxh, crh;
#pragma unroll
    for (int i = 0; i < 8; ++i) { cz[i] = bz; cr[i] = br; cxh[i] = bxh; crh[i] = brh; }

#pragma unroll
    for (int kb = 0; kb < 5; ++kb) {  // e @ Wc (K=160 padded)
      v16bf a = load_a_bf(sh_x, 160, kb, lane);
      cz  = wmma_bf16(a, load_b_frag(Wt, 5, kb, w,      lane), cz);
      cr  = wmma_bf16(a, load_b_frag(Wt, 5, kb, 13 + w, lane), cr);
      cxh = wmma_bf16(a, load_b_frag(Wt, 5, kb, 26 + w, lane), cxh);
    }
#pragma unroll
    for (int kb = 0; kb < 7; ++kb) {  // h @ Uc (K=224 padded)
      v16bf a = load_a_bf(sh_h, 224, kb, lane);
      cz  = wmma_bf16(a, load_b_frag(Ut, 7, kb, w,      lane), cz);
      cr  = wmma_bf16(a, load_b_frag(Ut, 7, kb, 13 + w, lane), cr);
      crh = wmma_bf16(a, load_b_frag(Ut, 7, kb, 26 + w, lane), crh);
    }
    __syncthreads();

#pragma unroll
    for (int i = 0; i < 8; ++i) {
      float z  = sigf(cz[i]);
      float r  = sigf(cr[i]);
      float hh = tanhf(cxh[i] + r * crh[i]);
      float h  = z * hfrag[i] + (1.f - z) * hh;
      hfrag[i] = h;
      int row = (lane < 16) ? i : i + 8;
      sh_h[row * 224 + col] = f2bf(h);
    }
  }
#pragma unroll
  for (int i = 0; i < 8; ++i) {
    int row = (lane < 16) ? i : i + 8;
    hc[(size_t)(bt * 16 + row) * 224 + col] = hfrag[i];
  }
}

// ---------------------------------------------------------------------------
// Generic dense layer: one wave per 16x16 output tile, WMMA k-loop.
// C[it,jt tile] = act(A @ Bpacked + bias)
// ---------------------------------------------------------------------------
__global__ void __launch_bounds__(32)
k_dense(const float* __restrict__ A, int lda, const bf16* __restrict__ Bp,
        int numKb, const float* __restrict__ bias, float* __restrict__ C,
        int ldc, int doRelu) {
  int it = blockIdx.x, jt = blockIdx.y;
  int lane = threadIdx.x, c16 = lane & 15;
  float bv = bias[jt * 16 + c16];
  v8f acc;
#pragma unroll
  for (int i = 0; i < 8; ++i) acc[i] = bv;
  const float* Arow = A + (size_t)it * 16 * lda;
  for (int kb = 0; kb < numKb; ++kb) {
    v16bf a = load_a_f32(Arow, lda, kb, lane);
    v16bf b = load_b_frag(Bp, numKb, kb, jt, lane);
    acc = wmma_bf16(a, b, acc);
  }
#pragma unroll
  for (int i = 0; i < 8; ++i) {
    float v = acc[i];
    if (doRelu) v = fmaxf(v, 0.f);
    int row = it * 16 + ((lane < 16) ? i : i + 8);
    C[(size_t)row * ldc + jt * 16 + c16] = v;
  }
}

// One wave per row, shuffle reductions (wave32).
__global__ void __launch_bounds__(32)
k_softmax(const float* __restrict__ src, int lds_, float* __restrict__ dst,
          int ldd, int nvalid) {
  int row = blockIdx.x, lane = threadIdx.x;
  const float* s = src + (size_t)row * lds_;
  float m = -1e30f;
  for (int c = lane; c < nvalid; c += 32) m = fmaxf(m, s[c]);
#pragma unroll
  for (int off = 16; off; off >>= 1) m = fmaxf(m, __shfl_xor(m, off, 32));
  float sum = 0.f;
  for (int c = lane; c < nvalid; c += 32) sum += __expf(s[c] - m);
#pragma unroll
  for (int off = 16; off; off >>= 1) sum += __shfl_xor(sum, off, 32);
  float inv = 1.f / sum;
  for (int c = lane; c < nvalid; c += 32)
    dst[(size_t)row * ldd + c] = __expf(s[c] - m) * inv;
}

// ---------------------------------------------------------------------------
extern "C" void kernel_launch(void* const* d_in, const int* in_sizes, int n_in,
                              void* d_out, int out_size, void* d_ws, size_t ws_size,
                              hipStream_t stream) {
  (void)in_sizes; (void)n_in; (void)out_size; (void)ws_size;
  const float* featx = (const float*)d_in[1];
  const int*   ci    = (const int*)d_in[2];
  const float* emb   = (const float*)d_in[3];
  const float* Wc    = (const float*)d_in[4];
  const float* Uc    = (const float*)d_in[5];
  const float* p_bc  = (const float*)d_in[6];
  const float* Wf    = (const float*)d_in[7];
  const float* Uf    = (const float*)d_in[8];
  const float* p_bf  = (const float*)d_in[9];
  const float* Wfd   = (const float*)d_in[10];
  const float* bfd   = (const float*)d_in[11];
  const float* Wc1   = (const float*)d_in[12];
  const float* bc1   = (const float*)d_in[13];
  const float* Wc2   = (const float*)d_in[14];
  const float* bc2   = (const float*)d_in[15];
  const float* W1    = (const float*)d_in[16];
  const float* b1    = (const float*)d_in[17];
  const float* W2    = (const float*)d_in[18];
  const float* b2    = (const float*)d_in[19];
  const float* Ws    = (const float*)d_in[20];
  const float* bs    = (const float*)d_in[21];

  char* base = (char*)d_ws;
  size_t off = 0;
  auto alloc = [&](size_t bytes) -> void* {
    void* p = base + off;
    off = (off + bytes + 255) & ~(size_t)255;
    return p;
  };
  bf16* WcP  = (bf16*)alloc(160 * 624 * 2);
  bf16* UcP  = (bf16*)alloc(224 * 624 * 2);
  bf16* WfP  = (bf16*)alloc(128 * 768 * 2);
  bf16* UfP  = (bf16*)alloc(256 * 768 * 2);
  bf16* WfdP = (bf16*)alloc(256 * 272 * 2);
  bf16* Wc1P = (bf16*)alloc(224 * 272 * 2);
  bf16* Wc2P = (bf16*)alloc(288 * 272 * 2);
  bf16* W1P  = (bf16*)alloc(544 * 272 * 2);
  bf16* W2P  = (bf16*)alloc(288 * 272 * 2);
  bf16* WsP  = (bf16*)alloc(288 * 16 * 2);
  float* bfdP = (float*)alloc(272 * 4);
  float* bc1P = (float*)alloc(272 * 4);
  float* bc2P = (float*)alloc(272 * 4);
  float* b1P  = (float*)alloc(272 * 4);
  float* b2P  = (float*)alloc(272 * 4);
  float* bsP  = (float*)alloc(16 * 4);
  float* act  = (float*)alloc((size_t)208 * 2176 * 4);
  float* hf = act;                  // [208][256]
  float* hc = hf + 208 * 256;       // [208][224]
  float* t1 = hc + 208 * 224;       // [208][288]
  float* l2 = t1 + 208 * 288;       // [208][272]
  float* xb = l2 + 208 * 272;       // [208][544] concat buffer
  float* x1 = xb + 208 * 544;       // [208][288]
  float* x2 = x1 + 208 * 288;       // [208][288]
  float* lg = x2 + 208 * 288;       // [208][16]

  auto pg = [](int total) { int g = (total + 255) / 256; return g > 1024 ? 1024 : g; };

  k_zero<<<512, 256, 0, stream>>>(act, 208 * 2176);

  k_pack<<<pg(160*624), 256, 0, stream>>>(Wc,  WcP,  150, 160, 3, 200, 208);
  k_pack<<<pg(224*624), 256, 0, stream>>>(Uc,  UcP,  200, 224, 3, 200, 208);
  k_pack<<<pg(128*768), 256, 0, stream>>>(Wf,  WfP,  128, 128, 3, 256, 256);
  k_pack<<<pg(256*768), 256, 0, stream>>>(Uf,  UfP,  256, 256, 3, 256, 256);
  k_pack<<<pg(256*272), 256, 0, stream>>>(Wfd, WfdP, 256, 256, 1, 264, 272);
  k_pack<<<pg(224*272), 256, 0, stream>>>(Wc1, Wc1P, 200, 224, 1, 264, 272);
  k_pack<<<pg(288*272), 256, 0, stream>>>(Wc2, Wc2P, 264, 288, 1, 264, 272);
  k_pack<<<pg(544*272), 256, 0, stream>>>(W1,  W1P,  528, 544, 1, 264, 272);
  k_pack<<<pg(288*272), 256, 0, stream>>>(W2,  W2P,  264, 288, 1, 264, 272);
  k_pack<<<pg(288*16),  256, 0, stream>>>(Ws,  WsP,  264, 288, 1, 8,   16);

  k_packbias<<<2, 256, 0, stream>>>(bfd, bfdP, 264, 272);
  k_packbias<<<2, 256, 0, stream>>>(bc1, bc1P, 264, 272);
  k_packbias<<<2, 256, 0, stream>>>(bc2, bc2P, 264, 272);
  k_packbias<<<2, 256, 0, stream>>>(b1,  b1P,  264, 272);
  k_packbias<<<2, 256, 0, stream>>>(b2,  b2P,  264, 272);
  k_packbias<<<1, 256, 0, stream>>>(bs,  bsP,  8,   16);

  // Persistent GRUs (batch-tile parallel; recurrence in registers)
  k_gru_feat<<<13, 512, 0, stream>>>(featx, WfP, UfP, p_bf, hf);
  k_gru_char<<<13, 416, 0, stream>>>(ci, emb, WcP, UcP, p_bc, hc);

  // feat_out = relu(hf @ Wfd + bfd) -> xb cols [0,272)
  k_dense<<<dim3(13, 17), 32, 0, stream>>>(hf, 256, WfdP, 256 / 32, bfdP, xb, 544, 1);
  // t1 = relu(hc @ Wc1 + bc1)
  k_dense<<<dim3(13, 17), 32, 0, stream>>>(hc, 224, Wc1P, 224 / 32, bc1P, t1, 288, 1);
  // l2 = t1 @ Wc2 + bc2
  k_dense<<<dim3(13, 17), 32, 0, stream>>>(t1, 288, Wc2P, 288 / 32, bc2P, l2, 272, 0);
  // char_out = softmax(l2) -> xb cols [264,528)  (runs after feat dense: overwrite ok)
  k_softmax<<<200, 32, 0, stream>>>(l2, 272, xb + 264, 544, 264);
  // x1 = relu(xb @ W1 + b1)   (xb cols 528..543 zero; W1 rows 528..543 zero)
  k_dense<<<dim3(13, 17), 32, 0, stream>>>(xb, 544, W1P, 544 / 32, b1P, x1, 288, 1);
  // x2 = relu(x1 @ W2 + b2)
  k_dense<<<dim3(13, 17), 32, 0, stream>>>(x1, 288, W2P, 288 / 32, b2P, x2, 288, 1);
  // logits = x2 @ Ws + bs
  k_dense<<<dim3(13, 1), 32, 0, stream>>>(x2, 288, WsP, 288 / 32, bsP, lg, 16, 0);
  // output softmax over 8 classes
  k_softmax<<<200, 32, 0, stream>>>(lg, 16, (float*)d_out, 8, 8);
}
